// EncoderLayer_2044404433629
// MI455X (gfx1250) — matrix operand order repaired and checked by reading, not compiled, not run on hardware
//
#include <hip/hip_runtime.h>

// MI455X (gfx1250) transformer encoder layer.
//  - every GEMM on v_wmma_f32_16x16x32_f16 (f16 in, f32 accum)
//  - weights pre-transposed ONCE so all staging is contiguous 16B chunks
//  - all A/B staging via GLOBAL_LOAD_ASYNC_TO_LDS_B128 (ASYNCcnt),
//    double-buffered to overlap DMA with WMMA
//  - flash attention (online softmax), V produced pre-transposed per head
//  - wave32 operand layouts per CDNA5 ISA 7.12.2

typedef __attribute__((ext_vector_type(16))) _Float16 v16h;
typedef __attribute__((ext_vector_type(8)))  _Float16 v8h;
typedef __attribute__((ext_vector_type(4)))  _Float16 v4h;
typedef __attribute__((ext_vector_type(8)))  float    v8f;

#define BB   4
#define SS   2048
#define DD   768
#define HH   12
#define DKK  64
#define DFF_ 3072
#define ROWS (BB * SS)   // 8192

typedef __attribute__((address_space(3))) _Float16 lds_f16_t;

// LDS byte offset of a __shared__ element (generic -> AS(3) -> int).
static __device__ __forceinline__ unsigned lds_off_of(_Float16* p) {
  return (unsigned)(unsigned long long)(lds_f16_t*)p;
}

// One 16-byte async DMA: global -> LDS. Tracked by ASYNCcnt.
static __device__ __forceinline__ void async_b128(unsigned lds_off,
                                                  const _Float16* g) {
  asm volatile("global_load_async_to_lds_b128 %0, %1, off"
               :: "v"(lds_off), "v"(g) : "memory");
}
static __device__ __forceinline__ void wait_async_le3() {
  asm volatile("s_wait_asynccnt 0x3" ::: "memory");
}
static __device__ __forceinline__ void wait_async_le2() {
  asm volatile("s_wait_asynccnt 0x2" ::: "memory");
}
static __device__ __forceinline__ void wait_async_0() {
  asm volatile("s_wait_asynccnt 0x0" ::: "memory");
}

// Build a 16-half WMMA operand from two b128 LDS loads.
static __device__ __forceinline__ v16h ld16(const _Float16* p) {
  v8h lo = *(const v8h*)p;
  v8h hi = *(const v8h*)(p + 8);
  return __builtin_shufflevector(lo, hi, 0, 1, 2, 3, 4, 5, 6, 7,
                                 8, 9, 10, 11, 12, 13, 14, 15);
}
// A-operand: two groups 16 halves apart.
static __device__ __forceinline__ v16h ld16_gap(const _Float16* p) {
  v8h lo = *(const v8h*)p;
  v8h hi = *(const v8h*)(p + 16);
  return __builtin_shufflevector(lo, hi, 0, 1, 2, 3, 4, 5, 6, 7,
                                 8, 9, 10, 11, 12, 13, 14, 15);
}

// ---------------------------------------------------------------- convert
__global__ void cvt_f32_f16_kernel(const float* __restrict__ in,
                                   _Float16* __restrict__ out, int n) {
  int i = (blockIdx.x * blockDim.x + threadIdx.x) * 4;
  if (i < n) {
    float4 f = *(const float4*)(in + i);
    v4h h = { (_Float16)f.x, (_Float16)f.y, (_Float16)f.z, (_Float16)f.w };
    *(v4h*)(out + i) = h;
  }
}

// ------------------------------------------------- one-time weight transpose
// W[K][N] f32 -> WT[N][K] f16, 32x32 LDS tiles (coalesced on both sides).
__global__ __launch_bounds__(256) void transpose_cvt_kernel(
    const float* __restrict__ W, _Float16* __restrict__ WT, int Kd, int Nd) {
  __shared__ _Float16 t[32][33];
  const int tid = threadIdx.x;
  const int k0 = blockIdx.y * 32, n0 = blockIdx.x * 32;
  {
    int r = tid >> 3, c = (tid & 7) << 2;
    float4 f = *(const float4*)(W + (size_t)(k0 + r) * Nd + n0 + c);
    t[r][c] = (_Float16)f.x; t[r][c + 1] = (_Float16)f.y;
    t[r][c + 2] = (_Float16)f.z; t[r][c + 3] = (_Float16)f.w;
  }
  __syncthreads();
  {
    int rn = tid >> 3, ck = (tid & 7) << 2;
    v4h p = { t[ck][rn], t[ck + 1][rn], t[ck + 2][rn], t[ck + 3][rn] };
    *(v4h*)(WT + (size_t)(n0 + rn) * Kd + k0 + ck) = p;
  }
}

// ---------------------------------------------------------------- GEMM
// C[M,N] = A[M,K] x Bt[N,K]^T + bias. Block 256 thr; macro-tile 128x64;
// wave (wr,wc) owns 32x32 (4 accumulators). Double-buffered async staging.
// outMode: 0 = f32 row-major, 1 = f16 row-major, 2 = f16 per-head transposed.
#define TM 128
#define TN 64
#define TK 32

__global__ __launch_bounds__(256) void gemm_f16_kernel(
    const _Float16* __restrict__ A, const _Float16* __restrict__ Bt,
    const float* __restrict__ bias, float* __restrict__ outF,
    _Float16* __restrict__ outH, int M, int N, int K,
    int outMode, int doRelu)
{
  __shared__ _Float16 lds_a[2][TM][TK + 8];   // stride 80B (16B aligned)
  __shared__ _Float16 lds_b[2][TN][TK + 8];

  const int tid  = threadIdx.x;
  const int lane = tid & 31;
  const int wid  = tid >> 5;
  const int wr   = wid & 3;      // rows wr*32
  const int wc   = wid >> 2;     // cols wc*32
  const int m0   = blockIdx.y * TM;
  const int n0   = blockIdx.x * TN;

  const int ar = tid >> 1;             // A row 0..127
  const int ac = (tid & 1) << 4;       // A col 0 or 16
  const int br = tid >> 2;             // Bt row (n) 0..63
  const int bc = (tid & 3) << 3;       // Bt col (k) 0,8,16,24

  v8f acc00 = {}, acc01 = {}, acc10 = {}, acc11 = {};

  auto issue = [&](int k0, int bufI) {
    const _Float16* ga = A + (size_t)(m0 + ar) * K + k0 + ac;
    unsigned la = lds_off_of(&lds_a[bufI][ar][ac]);
    async_b128(la, ga);
    async_b128(la + 16u, ga + 8);
    const _Float16* gb = Bt + (size_t)(n0 + br) * K + k0 + bc;
    async_b128(lds_off_of(&lds_b[bufI][br][bc]), gb);
    if (k0 + TK < K) __builtin_prefetch(ga + TK, 0, 3);
  };

  const int nch = K / TK;
  issue(0, 0);
  for (int ci = 0; ci < nch; ++ci) {
    if (ci + 1 < nch) { issue((ci + 1) * TK, (ci + 1) & 1); wait_async_le3(); }
    else              { wait_async_0(); }
    __syncthreads();

    const int bufI = ci & 1;
    const int am = lane & 15;
    const int half = lane >> 4;
    v16h a0 = ld16_gap(&lds_a[bufI][wr * 32 + am][half * 8]);
    v16h a1 = ld16_gap(&lds_a[bufI][wr * 32 + 16 + am][half * 8]);
    v16h b0 = ld16_gap(&lds_b[bufI][wc * 32 + am][half * 8]);
    v16h b1 = ld16_gap(&lds_b[bufI][wc * 32 + 16 + am][half * 8]);

    acc00 = __builtin_amdgcn_wmma_f32_16x16x32_f16(false, a0, false, b0, (short)0, acc00, false, false);
    acc01 = __builtin_amdgcn_wmma_f32_16x16x32_f16(false, a0, false, b1, (short)0, acc01, false, false);
    acc10 = __builtin_amdgcn_wmma_f32_16x16x32_f16(false, a1, false, b0, (short)0, acc10, false, false);
    acc11 = __builtin_amdgcn_wmma_f32_16x16x32_f16(false, a1, false, b1, (short)0, acc11, false, false);
    __syncthreads();
  }

  #pragma unroll
  for (int ti = 0; ti < 2; ++ti) {
    #pragma unroll
    for (int tj = 0; tj < 2; ++tj) {
      v8f acc = (ti == 0) ? (tj == 0 ? acc00 : acc01) : (tj == 0 ? acc10 : acc11);
      int n = n0 + wc * 32 + tj * 16 + (lane & 15);
      float bval = bias ? bias[n] : 0.0f;
      int mb = m0 + wr * 32 + ti * 16 + (lane >> 4) * 8;
      if (outMode == 2) {        // per-head transposed f16 (for V)
        int b_ = mb >> 11, s = mb & 2047;
        int h_ = n >> 6, dk = n & 63;
        v8h t;
        #pragma unroll
        for (int i = 0; i < 8; ++i) t[i] = (_Float16)(acc[i] + bval);
        *(v8h*)&outH[(((size_t)b_ * HH + h_) * DKK + dk) * SS + s] = t;
      } else {
        #pragma unroll
        for (int i = 0; i < 8; ++i) {
          float v = acc[i] + bval;
          if (doRelu) v = fmaxf(v, 0.0f);
          if (outMode == 0) outF[(size_t)(mb + i) * N + n] = v;
          else              outH[(size_t)(mb + i) * N + n] = (_Float16)v;
        }
      }
    }
  }
}

// ---------------------------------------------------------------- attention
__global__ __launch_bounds__(256) void attention_kernel(
    const _Float16* __restrict__ Q, const _Float16* __restrict__ K16,
    const _Float16* __restrict__ Vt, const int* __restrict__ mask,
    _Float16* __restrict__ ctx)
{
  __shared__ _Float16 ldsK[2][32][DKK + 8];   // [key][dk], stride 144B
  __shared__ _Float16 ldsV[2][DKK][32 + 8];   // [dk][key], stride 80B
  __shared__ _Float16 pbuf[8][16 * 32];       // per-wave P tile

  const int tid  = threadIdx.x;
  const int lane = tid & 31;
  const int wid  = tid >> 5;

  int blk = blockIdx.x;
  const int b    = blk / (HH * 16);
  int rem        = blk % (HH * 16);
  const int h    = rem / 16;
  const int qblk = rem % 16;
  const int q0   = (qblk * 8 + wid) * 16;

  const size_t rowbase = (size_t)b * SS;
  const int col0 = h * DKK;
  const _Float16* Vh = Vt + ((size_t)b * HH + h) * DKK * SS;

  const int skr = tid >> 3;            // K: key 0..31
  const int skc = (tid & 7) << 3;      // K: dk 0,8,...,56
  const int svd = tid >> 2;            // V: dk 0..63
  const int svk = (tid & 3) << 3;      // V: key 0,8,16,24

  auto issue = [&](int kb, int bufI) {
    async_b128(lds_off_of(&ldsK[bufI][skr][skc]),
               K16 + (rowbase + kb + skr) * DD + col0 + skc);
    async_b128(lds_off_of(&ldsV[bufI][svd][svk]),
               Vh + (size_t)svd * SS + kb + svk);
  };

  v16h aq0, aq1;
  {
    int m = lane & 15, half = lane >> 4;
    const _Float16* qr = Q + (rowbase + q0 + m) * DD + col0;
    aq0 = ld16_gap(qr + half * 8);
    aq1 = ld16_gap(qr + 32 + half * 8);
  }

  float m_run[8], l_run[8];
  #pragma unroll
  for (int i = 0; i < 8; ++i) { m_run[i] = -3.0e38f; l_run[i] = 0.0f; }
  v8f acc0 = {}, acc1 = {}, acc2 = {}, acc3 = {};

  _Float16* pw = &pbuf[wid][0];
  const int n  = lane & 15;
  const int kg = lane >> 4;

  issue(0, 0);
  const int nsteps = SS / 32;   // 64
  for (int it = 0; it < nsteps; ++it) {
    const int kb = it * 32;
    if (it + 1 < nsteps) { issue(kb + 32, (it + 1) & 1); wait_async_le2(); }
    else                 { wait_async_0(); }
    __syncthreads();
    const int bufI = it & 1;

    v8f s0 = {}, s1 = {};
    {
      v16h b00 = ld16_gap(&ldsK[bufI][n][kg * 8]);
      v16h b01 = ld16_gap(&ldsK[bufI][n][32 + kg * 8]);
      v16h b10 = ld16_gap(&ldsK[bufI][16 + n][kg * 8]);
      v16h b11 = ld16_gap(&ldsK[bufI][16 + n][32 + kg * 8]);
      s0 = __builtin_amdgcn_wmma_f32_16x16x32_f16(false, aq0, false, b00, (short)0, s0, false, false);
      s0 = __builtin_amdgcn_wmma_f32_16x16x32_f16(false, aq1, false, b01, (short)0, s0, false, false);
      s1 = __builtin_amdgcn_wmma_f32_16x16x32_f16(false, aq0, false, b10, (short)0, s1, false, false);
      s1 = __builtin_amdgcn_wmma_f32_16x16x32_f16(false, aq1, false, b11, (short)0, s1, false, false);
    }

    int mv0 = mask[rowbase + kb + n];
    int mv1 = mask[rowbase + kb + 16 + n];
    float p0[8], p1[8];
    #pragma unroll
    for (int i = 0; i < 8; ++i) {
      float v0 = mv0 ? s0[i] * 0.125f : -1.0e9f;
      float v1 = mv1 ? s1[i] * 0.125f : -1.0e9f;
      float bm = fmaxf(v0, v1);
      #pragma unroll
      for (int off = 1; off < 16; off <<= 1) bm = fmaxf(bm, __shfl_xor(bm, off));
      float mnew  = fmaxf(m_run[i], bm);
      float scale = __expf(m_run[i] - mnew);
      float e0 = __expf(v0 - mnew);
      float e1 = __expf(v1 - mnew);
      float bs = e0 + e1;
      #pragma unroll
      for (int off = 1; off < 16; off <<= 1) bs += __shfl_xor(bs, off);
      l_run[i] = l_run[i] * scale + bs;
      m_run[i] = mnew;
      p0[i] = e0; p1[i] = e1;
      acc0[i] *= scale; acc1[i] *= scale; acc2[i] *= scale; acc3[i] *= scale;
    }

    {
      int half = lane >> 4, c = lane & 15;
      #pragma unroll
      for (int i = 0; i < 8; ++i) {
        int r = half * 8 + i;
        pw[r * 32 + c]      = (_Float16)p0[i];
        pw[r * 32 + 16 + c] = (_Float16)p1[i];
      }
    }
    asm volatile("s_wait_dscnt 0x0" ::: "memory");
    v16h ap = ld16_gap(pw + (lane & 15) * 32 + (lane >> 4) * 8);

    {
      v16h bv0 = ld16_gap(&ldsV[bufI][n][kg * 8]);
      v16h bv1 = ld16_gap(&ldsV[bufI][16 + n][kg * 8]);
      v16h bv2 = ld16_gap(&ldsV[bufI][32 + n][kg * 8]);
      v16h bv3 = ld16_gap(&ldsV[bufI][48 + n][kg * 8]);
      acc0 = __builtin_amdgcn_wmma_f32_16x16x32_f16(false, ap, false, bv0, (short)0, acc0, false, false);
      acc1 = __builtin_amdgcn_wmma_f32_16x16x32_f16(false, ap, false, bv1, (short)0, acc1, false, false);
      acc2 = __builtin_amdgcn_wmma_f32_16x16x32_f16(false, ap, false, bv2, (short)0, acc2, false, false);
      acc3 = __builtin_amdgcn_wmma_f32_16x16x32_f16(false, ap, false, bv3, (short)0, acc3, false, false);
    }
    __syncthreads();
  }

  {
    int half = lane >> 4;
    #pragma unroll
    for (int i = 0; i < 8; ++i) {
      int row = q0 + half * 8 + i;
      float invl = 1.0f / l_run[i];
      _Float16* cr = ctx + (rowbase + row) * DD + col0;
      cr[n]      = (_Float16)(acc0[i] * invl);
      cr[16 + n] = (_Float16)(acc1[i] * invl);
      cr[32 + n] = (_Float16)(acc2[i] * invl);
      cr[48 + n] = (_Float16)(acc3[i] * invl);
    }
  }
}

// ---------------------------------------------------------------- add + LN
__global__ __launch_bounds__(256) void add_ln_kernel(
    const float* __restrict__ x_in, const float* __restrict__ add_in,
    const float* __restrict__ gamma, const float* __restrict__ beta,
    float* __restrict__ outF, _Float16* __restrict__ outH)
{
  __shared__ float red[256];
  const int row = blockIdx.x;
  const int tid = threadIdx.x;
  const float* xr = x_in  + (size_t)row * DD;
  const float* ar = add_in + (size_t)row * DD;

  float v[3];
  float s = 0.0f;
  #pragma unroll
  for (int j = 0; j < 3; ++j) { v[j] = xr[tid + j * 256] + ar[tid + j * 256]; s += v[j]; }

  red[tid] = s; __syncthreads();
  for (int off = 128; off > 0; off >>= 1) { if (tid < off) red[tid] += red[tid + off]; __syncthreads(); }
  float mu = red[0] * (1.0f / DD);
  __syncthreads();

  float s2 = 0.0f;
  #pragma unroll
  for (int j = 0; j < 3; ++j) { float d = v[j] - mu; s2 += d * d; }
  red[tid] = s2; __syncthreads();
  for (int off = 128; off > 0; off >>= 1) { if (tid < off) red[tid] += red[tid + off]; __syncthreads(); }
  float inv = rsqrtf(red[0] * (1.0f / DD) + 1e-5f);

  #pragma unroll
  for (int j = 0; j < 3; ++j) {
    int c = tid + j * 256;
    float y = (v[j] - mu) * inv * gamma[c] + beta[c];
    if (outF) outF[(size_t)row * DD + c] = y;
    if (outH) outH[(size_t)row * DD + c] = (_Float16)y;
  }
}

// ---------------------------------------------------------------- driver
extern "C" void kernel_launch(void* const* d_in, const int* in_sizes, int n_in,
                              void* d_out, int out_size, void* d_ws, size_t ws_size,
                              hipStream_t stream)
{
  (void)in_sizes; (void)n_in; (void)out_size; (void)ws_size;

  const float* src  = (const float*)d_in[0];
  const int*   mask = (const int*)  d_in[1];
  const float* Wq   = (const float*)d_in[2];
  const float* bq   = (const float*)d_in[3];
  const float* Wk   = (const float*)d_in[4];
  const float* bk   = (const float*)d_in[5];
  const float* Wv   = (const float*)d_in[6];
  const float* bv   = (const float*)d_in[7];
  const float* Wo   = (const float*)d_in[8];
  const float* bo   = (const float*)d_in[9];
  const float* W1   = (const float*)d_in[10];
  const float* b1   = (const float*)d_in[11];
  const float* W2   = (const float*)d_in[12];
  const float* b2   = (const float*)d_in[13];
  const float* ln1g = (const float*)d_in[14];
  const float* ln1b = (const float*)d_in[15];
  const float* ln2g = (const float*)d_in[16];
  const float* ln2b = (const float*)d_in[17];

  char* ws = (char*)d_ws;
  const size_t SZ_X16 = (size_t)ROWS * DD * 2;
  const size_t SZ_W   = (size_t)DD * DD * 2;
  const size_t SZ_W1  = (size_t)DD * DFF_ * 2;
  const size_t SZ_F32 = (size_t)ROWS * DD * 4;

  size_t o = 0;
  _Float16* X16  = (_Float16*)(ws + o); o += SZ_X16;
  _Float16* WqT  = (_Float16*)(ws + o); o += SZ_W;
  _Float16* WkT  = (_Float16*)(ws + o); o += SZ_W;
  _Float16* WvT  = (_Float16*)(ws + o); o += SZ_W;
  _Float16* WoT  = (_Float16*)(ws + o); o += SZ_W;
  _Float16* W1T  = (_Float16*)(ws + o); o += SZ_W1;
  _Float16* W2T  = (_Float16*)(ws + o); o += SZ_W1;
  _Float16* Q16  = (_Float16*)(ws + o); o += SZ_X16;
  _Float16* K16  = (_Float16*)(ws + o); o += SZ_X16;
  _Float16* Vt16 = (_Float16*)(ws + o); o += SZ_X16;
  _Float16* C16  = (_Float16*)(ws + o); o += SZ_X16;
  float*    attnF= (float*)   (ws + o); o += SZ_F32;
  float*    xF   = (float*)   (ws + o); o += SZ_F32;
  _Float16* x16  = (_Float16*)(ws + o); o += SZ_X16;
  _Float16* ffnh = Q16;
  float*    ffnF = attnF;

  cvt_f32_f16_kernel<<<(ROWS * DD / 4 + 255) / 256, 256, 0, stream>>>(src, X16, ROWS * DD);
  transpose_cvt_kernel<<<dim3(DD / 32, DD / 32),   256, 0, stream>>>(Wq, WqT, DD, DD);
  transpose_cvt_kernel<<<dim3(DD / 32, DD / 32),   256, 0, stream>>>(Wk, WkT, DD, DD);
  transpose_cvt_kernel<<<dim3(DD / 32, DD / 32),   256, 0, stream>>>(Wv, WvT, DD, DD);
  transpose_cvt_kernel<<<dim3(DD / 32, DD / 32),   256, 0, stream>>>(Wo, WoT, DD, DD);
  transpose_cvt_kernel<<<dim3(DFF_ / 32, DD / 32), 256, 0, stream>>>(W1, W1T, DD, DFF_);
  transpose_cvt_kernel<<<dim3(DD / 32, DFF_ / 32), 256, 0, stream>>>(W2, W2T, DFF_, DD);

  dim3 blk(256);
  dim3 gD(DD / TN, ROWS / TM);      // 12 x 64
  dim3 gF(DFF_ / TN, ROWS / TM);    // 48 x 64

  gemm_f16_kernel<<<gD, blk, 0, stream>>>(X16, WqT, bq, nullptr, Q16,  ROWS, DD, DD, 1, 0);
  gemm_f16_kernel<<<gD, blk, 0, stream>>>(X16, WkT, bk, nullptr, K16,  ROWS, DD, DD, 1, 0);
  gemm_f16_kernel<<<gD, blk, 0, stream>>>(X16, WvT, bv, nullptr, Vt16, ROWS, DD, DD, 2, 0);

  attention_kernel<<<BB * HH * 16, blk, 0, stream>>>(Q16, K16, Vt16, mask, C16);

  gemm_f16_kernel<<<gD, blk, 0, stream>>>(C16, WoT, bo, attnF, nullptr, ROWS, DD, DD, 0, 0);
  add_ln_kernel<<<ROWS, blk, 0, stream>>>(src, attnF, ln1g, ln1b, xF, x16);

  gemm_f16_kernel<<<gF, blk, 0, stream>>>(x16, W1T, b1, nullptr, ffnh, ROWS, DFF_, DD, 1, 1);
  gemm_f16_kernel<<<gD, blk, 0, stream>>>(ffnh, W2T, b2, ffnF, nullptr, ROWS, DD, DFF_, 0, 0);

  add_ln_kernel<<<ROWS, blk, 0, stream>>>(xF, ffnF, ln2g, ln2b, (float*)d_out, nullptr);
}
